// NormalizedNNN_1486058684851
// MI455X (gfx1250) — matrix operand order repaired and checked by reading, not compiled, not run on hardware
//
#include <hip/hip_runtime.h>
#include <hip/hip_bf16.h>
#include <cstdint>
#include <cstddef>

typedef _Float16 v16h __attribute__((ext_vector_type(16)));
typedef float    v8f  __attribute__((ext_vector_type(8)));
typedef unsigned int u32x4 __attribute__((ext_vector_type(4)));
typedef int          i32x8 __attribute__((ext_vector_type(8)));
typedef int          i32x4 __attribute__((ext_vector_type(4)));

#define EPSN 1e-20f

__device__ __forceinline__ float softplusf(float x) {
  return (x > 20.f) ? x : log1pf(expf(x));
}

// ---------------------------------------------------------------------------
// TDM: load an 18-row x 576-element (f16) 2D tile from padded-NHWC global
// memory into LDS. With interleave=true, D# LDS padding (16 DWORDs every
// 16 DWORDs) interleaves this 32-channel chunk with the other chunk so the
// LDS ends up pixel-major channel-last with CIN=64.
// D# layout per CDNA5 ISA ch.8 (group0: count/lds_addr/global_addr/type,
// group1: data_size + dims/strides/pad). 6-arg clang-23 builtin.
// ---------------------------------------------------------------------------
__device__ __forceinline__ void tdm_load_tile(unsigned lds_byte_off, const void* gsrc,
                                              int src_stride_elems, bool interleave) {
  const unsigned ROWLEN = 18 * 32;   // elements per tile row (contiguous in src)
  const unsigned ROWS   = 18;
  unsigned long long ga = (unsigned long long)(uintptr_t)gsrc;
  u32x4 g0;
  g0[0] = 1u;                                            // count=1, user mode
  g0[1] = lds_byte_off;                                  // lds_addr
  g0[2] = (unsigned)ga;                                  // global_addr[31:0]
  g0[3] = (unsigned)((ga >> 32) & 0x01FFFFFFu) | (2u << 30);  // addr[56:32] | type=2
  unsigned w0 = (1u << 16);                              // data_size=1 (2 bytes)
  if (interleave)                                        // pad 16DW after every 16DW
    w0 |= (1u << 20) | (3u << 22) | (15u << 25);         // pad_en | interval=16DW | amount=16DW
  i32x8 g1;
  g1[0] = (int)w0;
  g1[1] = (int)(ROWLEN << 16);                           // tensor_dim0[15:0] in [63:48]
  g1[2] = (int)(ROWS << 16);                             // td0 hi=0 | tensor_dim1[15:0]
  g1[3] = (int)(ROWLEN << 16);                           // td1 hi=0 | tile_dim0
  g1[4] = (int)ROWS;                                     // tile_dim1 | tile_dim2=0
  g1[5] = src_stride_elems;                              // tensor_dim0_stride[31:0]
  g1[6] = 0;                                             // stride hi | dim1_stride lo
  g1[7] = 0;
  i32x4 z4 = {0, 0, 0, 0};
  i32x8 z8 = {0, 0, 0, 0, 0, 0, 0, 0};
  __builtin_amdgcn_tensor_load_to_lds(g0, g1, z4, z4, z8, 0);
}

// ---------------------------------------------------------------------------
// Weight prep: softplus(W) -> f16, pre-swizzled into this code's WMMA A-matrix
// per-lane layout (16-bit A 16x32: lane L holds row M=L&15; half h -> K =
// 16*(L>>4) + h, contiguous k order, not the ISA table order; B loads use the same).
// Buffer layout: wp[((tap*KC+kc)*2 + mtile)*512 + lane*16 + h]
// ---------------------------------------------------------------------------
__global__ void prep_w_kernel(const float* __restrict__ Wsrc, _Float16* __restrict__ wp,
                              int csrc, int KC) {
  int tid = blockIdx.x * blockDim.x + threadIdx.x;
  int total = 9 * KC * 2 * 32;
  if (tid >= total) return;
  int lane = tid & 31;
  int mt   = (tid >> 5) & 1;
  int rest = tid >> 6;
  int kc  = rest % KC;
  int tap = rest / KC;
  int side = lane >> 4;
  int m    = lane & 15;
  int o = mt * 16 + m;
  int ob = ((tap * KC + kc) * 2 + mt) * 512 + lane * 16;
#pragma unroll
  for (int h = 0; h < 16; ++h) {
    int v = h >> 1, p = h & 1;
    int K = 16 * side + h;
    int i = kc * 32 + K;
    float val = 0.f;
    if (i < csrc) val = softplusf(Wsrc[(o * csrc + i) * 9 + tap]);
    wp[ob + h] = (_Float16)val;
  }
}

__global__ void wsum_kernel(const float* __restrict__ Wsrc, float* __restrict__ s, int csrc) {
  int o = threadIdx.x;
  if (o >= 32) return;
  float acc = 0.f;
  for (int i = 0; i < csrc * 9; ++i) acc += softplusf(Wsrc[o * csrc * 9 + i]);
  s[o] = acc;
}

// Zero the 1-pixel border of a padded-NHWC f16 buffer (provides conv zero-pad
// and makes upsample reads of the halo land on zeros).
__global__ void zero_border_kernel(_Float16* __restrict__ p, int H, int W) {
  int Hp = H + 2, Wp = W + 2;
  int nb = 2 * Wp + 2 * H;                // border pixels per batch image
  size_t total = (size_t)4 * nb * 4;      // 4 x uint4 (32 halves) per pixel
  size_t idx = (size_t)blockIdx.x * blockDim.x + threadIdx.x;
  if (idx >= total) return;
  int k = (int)(idx & 3);
  size_t t = idx >> 2;
  int pi = (int)(t % nb);
  int b  = (int)(t / nb);
  int y, x;
  if (pi < Wp)            { y = 0;      x = pi; }
  else if (pi < 2 * Wp)   { y = Hp - 1; x = pi - Wp; }
  else { int r = pi - 2 * Wp; y = 1 + (r >> 1); x = (r & 1) ? (Wp - 1) : 0; }
  uint4 z = {0u, 0u, 0u, 0u};
  *(uint4*)(p + ((((size_t)b * Hp + y) * Wp + x) << 5) + (k << 3)) = z;
}

// f32 NCHW 3-ch inputs -> padded NHWC f16 32-ch (channels 3..31 zero)
__global__ void convert_in_kernel(const float* __restrict__ x0, const float* __restrict__ c0,
                                  _Float16* __restrict__ X, _Float16* __restrict__ C,
                                  int H, int W) {
  size_t total = (size_t)4 * H * W * 32;
  size_t idx = (size_t)blockIdx.x * blockDim.x + threadIdx.x;
  if (idx >= total) return;
  int ch = (int)(idx & 31);
  size_t t = idx >> 5;
  int gx = (int)(t % W); t /= W;
  int gy = (int)(t % H); int b = (int)(t / H);
  float xv = 0.f, cv = 0.f;
  if (ch < 3) {
    size_t si = (((size_t)b * 3 + ch) * H + gy) * W + gx;
    xv = x0[si]; cv = c0[si];
  }
  size_t o = ((((size_t)b * (H + 2) + gy + 1) * (W + 2) + gx + 1) << 5) + ch;
  X[o] = (_Float16)xv;
  C[o] = (_Float16)cv;
}

// ---------------------------------------------------------------------------
// Normalized 3x3 conv via WMMA implicit GEMM. Activations: padded NHWC f16.
// Block = 16x16 output pixels, 256 threads (8 wave32's), batch = blockIdx.z.
// Chunk A = channels 0..31, chunk B = 32..63 (KC==2). Per-pointer shifts fold
// nearest-upsample + concat (incl. the swapped (c23,c1) final pairing) into
// the staging address math. Shift-0 arrays stage via TDM tensor loads.
// ---------------------------------------------------------------------------
template<int KC, bool F32OUT>
__global__ __launch_bounds__(256)
void nconv_wmma_kernel(const _Float16* __restrict__ xA, const _Float16* __restrict__ cA,
                       int shAx, int shAc,
                       const _Float16* __restrict__ xB, const _Float16* __restrict__ cB,
                       int shBx, int shBc,
                       const _Float16* __restrict__ wp,
                       const float* __restrict__ bias,
                       const float* __restrict__ wsum,
                       void* __restrict__ xout, void* __restrict__ cout,
                       int H, int W) {
  constexpr int CIN  = 32 * KC;
  constexpr int NPIX = 18 * 18;
  __shared__ __align__(32) _Float16 sX[NPIX * CIN];   // x   (f16, pixel-major channel-last)
  __shared__ __align__(32) _Float16 sC[NPIX * CIN];   // c

  const int tid = threadIdx.x;
  const int b   = blockIdx.z;
  const int ty0 = blockIdx.y * 16, tx0 = blockIdx.x * 16;

  // ---- Stage 18x18 halo tile: TDM block copy (shift 0) or uint4 gather ----
#pragma unroll
  for (int q = 0; q < KC; ++q) {
    const _Float16* srX = q ? xB : xA; const int sx = q ? shBx : shAx;
    const _Float16* srC = q ? cB : cA; const int sc = q ? shBc : shAc;
    // X array of this chunk
    if (sx == 0) {
      if (tid < 32) {
        const _Float16* g = srX + ((((size_t)b * (H + 2) + ty0) * (W + 2) + tx0) << 5);
        tdm_load_tile((unsigned)(uintptr_t)sX + q * 64, g, (W + 2) * 32, KC == 2);
      }
    } else {
      for (int idx = tid; idx < NPIX * 4; idx += 256) {
        int pix = idx >> 2, k = idx & 3;
        int py = pix / 18, px = pix - py * 18;
        int gy = ty0 - 1 + py, gx = tx0 - 1 + px;
        int hs = H >> sx, wsd = W >> sx;
        size_t so = ((((size_t)b * (hs + 2) + ((gy >> sx) + 1)) * (wsd + 2) + ((gx >> sx) + 1)) << 5) + (k << 3);
        *(uint4*)&sX[(size_t)pix * CIN + q * 32 + k * 8] = *(const uint4*)(srX + so);
      }
    }
    // C array of this chunk
    if (sc == 0) {
      if (tid < 32) {
        const _Float16* g = srC + ((((size_t)b * (H + 2) + ty0) * (W + 2) + tx0) << 5);
        tdm_load_tile((unsigned)(uintptr_t)sC + q * 64, g, (W + 2) * 32, KC == 2);
      }
    } else {
      for (int idx = tid; idx < NPIX * 4; idx += 256) {
        int pix = idx >> 2, k = idx & 3;
        int py = pix / 18, px = pix - py * 18;
        int gy = ty0 - 1 + py, gx = tx0 - 1 + px;
        int hs = H >> sc, wsd = W >> sc;
        size_t so = ((((size_t)b * (hs + 2) + ((gy >> sc) + 1)) * (wsd + 2) + ((gx >> sc) + 1)) << 5) + (k << 3);
        *(uint4*)&sC[(size_t)pix * CIN + q * 32 + k * 8] = *(const uint4*)(srC + so);
      }
    }
  }
  if (tid < 32) __builtin_amdgcn_s_wait_tensorcnt((short)0);
  __syncthreads();

  const int wave = tid >> 5;
  const int lane = tid & 31;
  const int side = lane >> 4;
  const int m    = lane & 15;

  for (int rr = 0; rr < 2; ++rr) {
    const int row = wave * 2 + rr;
    v8f ax0 = {}, ax1 = {}, ac0 = {}, ac1 = {};
#pragma unroll
    for (int kc = 0; kc < KC; ++kc) {
#pragma unroll
      for (int tap = 0; tap < 9; ++tap) {
        const int dy = tap / 3, dx = tap % 3;
        // B operand (32x16 f16): lane = pixel N=m, K = 16*side + h
        const int bo = ((row + dy) * 18 + (m + dx)) * CIN + kc * 32 + side * 16;
        v16h Bx = *(const v16h*)(sX + bo);
        v16h Bc = *(const v16h*)(sC + bo);
        v16h Ba = Bx * Bc;                       // a = x*c via v_pk_mul_f16
        const _Float16* ap = wp + (size_t)((tap * KC + kc) * 2) * 512;
        v16h A0 = *(const v16h*)(ap + lane * 16);
        v16h A1 = *(const v16h*)(ap + 512 + lane * 16);
        ax0 = __builtin_amdgcn_wmma_f32_16x16x32_f16(false, A0, false, Ba, (short)0, ax0, false, false);
        ac0 = __builtin_amdgcn_wmma_f32_16x16x32_f16(false, A0, false, Bc, (short)0, ac0, false, false);
        ax1 = __builtin_amdgcn_wmma_f32_16x16x32_f16(false, A1, false, Ba, (short)0, ax1, false, false);
        ac1 = __builtin_amdgcn_wmma_f32_16x16x32_f16(false, A1, false, Bc, (short)0, ac1, false, false);
      }
    }
    // ---- Epilogue: xo = nomin/(denom+eps)+b ; co = denom / sum(w) ----
    const int gy = ty0 + row;
    const int gx = tx0 + m;
    if (F32OUT) {
      float* xo = (float*)xout;
      float* co = (float*)cout;
      const size_t bstr = (size_t)H * W;
      const size_t po = (size_t)gy * W + gx;
#pragma unroll
      for (int r = 0; r < 8; ++r) {
        int oc = r + side * 8;                  // D layout: M = r + 8*(L>>4), N = L&15
        float d0 = ac0[r];
        xo[((size_t)b * 32 + oc) * bstr + po] = ax0[r] / (d0 + EPSN) + bias[oc];
        co[((size_t)b * 32 + oc) * bstr + po] = d0 / wsum[oc];
        int oc1 = oc + 16;
        float d1 = ac1[r];
        xo[((size_t)b * 32 + oc1) * bstr + po] = ax1[r] / (d1 + EPSN) + bias[oc1];
        co[((size_t)b * 32 + oc1) * bstr + po] = d1 / wsum[oc1];
      }
    } else {
      _Float16* xo = (_Float16*)xout;
      _Float16* co = (_Float16*)cout;
      const size_t pb = ((((size_t)b * (H + 2) + gy + 1) * (W + 2) + gx + 1) << 5);
#pragma unroll
      for (int r = 0; r < 8; ++r) {
        int oc = r + side * 8;
        float d0 = ac0[r];
        xo[pb + oc] = (_Float16)(ax0[r] / (d0 + EPSN) + bias[oc]);
        co[pb + oc] = (_Float16)(d0 / wsum[oc]);
        int oc1 = oc + 16;
        float d1 = ac1[r];
        xo[pb + oc1] = (_Float16)(ax1[r] / (d1 + EPSN) + bias[oc1]);
        co[pb + oc1] = (_Float16)(d1 / wsum[oc1]);
      }
    }
  }
}

// 2x2 max-pool on c with argmax-gather of x; c/=4. First-occurrence tie-break
// in (ysub,xsub) order matches jnp.argmax. Padded NHWC f16 in and out.
__global__ void pool_gather_kernel(const _Float16* __restrict__ x, const _Float16* __restrict__ c,
                                   _Float16* __restrict__ xo, _Float16* __restrict__ co,
                                   int H, int W) {
  int H2 = H >> 1, W2 = W >> 1;
  int Hp = H + 2, Wp = W + 2, Wp2 = W2 + 2;
  size_t total = (size_t)4 * H2 * W2 * 32;
  size_t idx = (size_t)blockIdx.x * blockDim.x + threadIdx.x;
  if (idx >= total) return;
  int ch = (int)(idx & 31);
  size_t t = idx >> 5;
  int gx2 = (int)(t % W2); t /= W2;
  int gy2 = (int)(t % H2); int b = (int)(t / H2);
  size_t rb = ((((size_t)b * Hp + 2 * gy2 + 1) * Wp + 2 * gx2 + 1) << 5) + ch;
  size_t rs = (size_t)Wp << 5;
  float c00 = (float)c[rb],      c01 = (float)c[rb + 32];
  float c10 = (float)c[rb + rs], c11 = (float)c[rb + rs + 32];
  int best = 0; float bc = c00;
  if (c01 > bc) { bc = c01; best = 1; }
  if (c10 > bc) { bc = c10; best = 2; }
  if (c11 > bc) { bc = c11; best = 3; }
  _Float16 bx = x[rb + (size_t)(best >> 1) * rs + (size_t)(best & 1) * 32];
  size_t ob = ((((size_t)b * (H2 + 2) + gy2 + 1) * Wp2 + gx2 + 1) << 5) + ch;
  xo[ob] = bx;
  co[ob] = (_Float16)(bc * 0.25f);
}

// Fused final 1x1 convs: out = W2*(W1*x + b1) + b2  (32 -> 3 -> 3), f32 NCHW in
__global__ void conv1x1_fused_kernel(const float* __restrict__ xin,
                                     const float* __restrict__ W1, const float* __restrict__ b1,
                                     const float* __restrict__ W2, const float* __restrict__ b2,
                                     float* __restrict__ out, int HW) {
  int idx = blockIdx.x * blockDim.x + threadIdx.x;
  int total = 4 * HW;
  if (idx >= total) return;
  int b = idx / HW;
  int pix = idx - b * HW;
  float t0 = b1[0], t1 = b1[1], t2 = b1[2];
  const float* xp = xin + (size_t)b * 32 * HW + pix;
#pragma unroll 8
  for (int i = 0; i < 32; ++i) {
    float v = xp[(size_t)i * HW];
    t0 += W1[i] * v; t1 += W1[32 + i] * v; t2 += W1[64 + i] * v;
  }
  float* op = out + (size_t)b * 3 * HW + pix;
  op[0]              = b2[0] + W2[0] * t0 + W2[1] * t1 + W2[2] * t2;
  op[(size_t)HW]     = b2[1] + W2[3] * t0 + W2[4] * t1 + W2[5] * t2;
  op[(size_t)2 * HW] = b2[2] + W2[6] * t0 + W2[7] * t1 + W2[8] * t2;
}

// ---------------------------------------------------------------------------
extern "C" void kernel_launch(void* const* d_in, const int* in_sizes, int n_in,
                              void* d_out, int out_size, void* d_ws, size_t ws_size,
                              hipStream_t stream) {
  (void)in_sizes; (void)n_in; (void)out_size; (void)ws_size;
  const float* x0  = (const float*)d_in[0];
  const float* c0  = (const float*)d_in[1];
  const float* Wd1 = (const float*)d_in[2];  const float* bd1 = (const float*)d_in[3];
  const float* Wd2 = (const float*)d_in[4];  const float* bd2 = (const float*)d_in[5];
  const float* Wd3 = (const float*)d_in[6];  const float* bd3 = (const float*)d_in[7];
  const float* Wu1 = (const float*)d_in[8];  const float* bu1 = (const float*)d_in[9];
  const float* Wu2 = (const float*)d_in[10]; const float* bu2 = (const float*)d_in[11];
  const float* Wu3 = (const float*)d_in[12]; const float* bu3 = (const float*)d_in[13];
  const float* Wc1 = (const float*)d_in[14]; const float* bc1 = (const float*)d_in[15];
  const float* Wc2 = (const float*)d_in[16]; const float* bc2 = (const float*)d_in[17];

  const int B = 4, H = 512, W = 512;
  auto PH = [](int h) { return (size_t)4 * (h + 2) * (h + 2) * 32; };  // padded NHWC halves
  const size_t P512 = PH(512), P256 = PH(256), P128 = PH(128), P64 = PH(64);
  const size_t S512 = (size_t)B * 32 * H * W;

  char* base = (char*)d_ws;
  size_t off = 0;
  auto allocf = [&](size_t n) -> float* {
    off = (off + 255) & ~(size_t)255;
    float* p = (float*)(base + off);
    off += n * sizeof(float);
    return p;
  };
  auto alloch = [&](size_t n) -> _Float16* {
    off = (off + 255) & ~(size_t)255;
    _Float16* p = (_Float16*)(base + off);
    off += n * sizeof(_Float16);
    return p;
  };

  _Float16* Xin = alloch(P512); _Float16* Cin = alloch(P512);
  _Float16* F0x = alloch(P512); _Float16* F0c = alloch(P512);   // x1,c1 persist
  _Float16* F1x = alloch(P512); _Float16* F1c = alloch(P512);
  _Float16* L1Ax = alloch(P256); _Float16* L1Ac = alloch(P256);
  _Float16* L1Bx = alloch(P256); _Float16* L1Bc = alloch(P256);
  _Float16* L2Ax = alloch(P128); _Float16* L2Ac = alloch(P128); // x2d,c2d persist
  _Float16* L2Bx = alloch(P128); _Float16* L2Bc = alloch(P128); // x23d,c23d
  _Float16* L3Ax = alloch(P64);  _Float16* L3Ac = alloch(P64);  // x3d,c3d
  _Float16* L3Bx = alloch(P64);  _Float16* L3Bc = alloch(P64);  // x34d,c34d
  float* Xpre = allocf(S512);                                   // final nconv x (f32 NCHW)
  _Float16* wpD1 = alloch(9216);  _Float16* wpD2 = alloch(9216);  _Float16* wpD3 = alloch(9216);
  _Float16* wpU1 = alloch(18432); _Float16* wpU2 = alloch(18432); _Float16* wpU3 = alloch(18432);
  float* sD1 = allocf(32); float* sD2 = allocf(32); float* sD3 = allocf(32);
  float* sU1 = allocf(32); float* sU2 = allocf(32); float* sU3 = allocf(32);

  // --- weight prep ---
  prep_w_kernel<<<dim3(9), 64, 0, stream>>>(Wd1, wpD1, 3, 1);
  prep_w_kernel<<<dim3(9), 64, 0, stream>>>(Wd2, wpD2, 32, 1);
  prep_w_kernel<<<dim3(9), 64, 0, stream>>>(Wd3, wpD3, 32, 1);
  prep_w_kernel<<<dim3(18), 64, 0, stream>>>(Wu1, wpU1, 64, 2);
  prep_w_kernel<<<dim3(18), 64, 0, stream>>>(Wu2, wpU2, 64, 2);
  prep_w_kernel<<<dim3(18), 64, 0, stream>>>(Wu3, wpU3, 64, 2);
  wsum_kernel<<<1, 32, 0, stream>>>(Wd1, sD1, 3);
  wsum_kernel<<<1, 32, 0, stream>>>(Wd2, sD2, 32);
  wsum_kernel<<<1, 32, 0, stream>>>(Wd3, sD3, 32);
  wsum_kernel<<<1, 32, 0, stream>>>(Wu1, sU1, 64);
  wsum_kernel<<<1, 32, 0, stream>>>(Wu2, sU2, 64);
  wsum_kernel<<<1, 32, 0, stream>>>(Wu3, sU3, 64);

  // --- zero padded borders (deterministic each call) ---
  auto zb = [&](_Float16* p, int h) {
    size_t n = (size_t)4 * (2 * (h + 2) + 2 * h) * 4;
    zero_border_kernel<<<dim3((unsigned)((n + 255) / 256)), 256, 0, stream>>>(p, h, h);
  };
  zb(Xin, 512); zb(Cin, 512); zb(F0x, 512); zb(F0c, 512); zb(F1x, 512); zb(F1c, 512);
  zb(L1Ax, 256); zb(L1Ac, 256); zb(L1Bx, 256); zb(L1Bc, 256);
  zb(L2Ax, 128); zb(L2Ac, 128); zb(L2Bx, 128); zb(L2Bc, 128);
  zb(L3Ax, 64);  zb(L3Ac, 64);  zb(L3Bx, 64);  zb(L3Bc, 64);

  // --- input conversion: f32 NCHW 3ch -> padded NHWC f16 32ch ---
  {
    size_t n = (size_t)4 * 512 * 512 * 32;
    convert_in_kernel<<<dim3((unsigned)((n + 255) / 256)), 256, 0, stream>>>(x0, c0, Xin, Cin, 512, 512);
  }

  dim3 g512(32, 32, B), g256(16, 16, B), g128(8, 8, B), g64(4, 4, B);

  // --- encoder, full res ---
  nconv_wmma_kernel<1, false><<<g512, 256, 0, stream>>>(Xin, Cin, 0, 0, nullptr, nullptr, 0, 0,
                                                        wpD1, bd1, sD1, F0x, F0c, 512, 512);
  nconv_wmma_kernel<1, false><<<g512, 256, 0, stream>>>(F0x, F0c, 0, 0, nullptr, nullptr, 0, 0,
                                                        wpD2, bd2, sD2, F1x, F1c, 512, 512);
  nconv_wmma_kernel<1, false><<<g512, 256, 0, stream>>>(F1x, F1c, 0, 0, nullptr, nullptr, 0, 0,
                                                        wpD3, bd3, sD3, F0x, F0c, 512, 512); // x1,c1
  // --- level 1 (256) ---
  {
    size_t n = (size_t)4 * 256 * 256 * 32;
    pool_gather_kernel<<<dim3((unsigned)((n + 255) / 256)), 256, 0, stream>>>(F0x, F0c, L1Ax, L1Ac, 512, 512);
  }
  nconv_wmma_kernel<1, false><<<g256, 256, 0, stream>>>(L1Ax, L1Ac, 0, 0, nullptr, nullptr, 0, 0,
                                                        wpD2, bd2, sD2, L1Bx, L1Bc, 256, 256);
  nconv_wmma_kernel<1, false><<<g256, 256, 0, stream>>>(L1Bx, L1Bc, 0, 0, nullptr, nullptr, 0, 0,
                                                        wpD3, bd3, sD3, L1Ax, L1Ac, 256, 256);
  // --- level 2 (128) ---
  {
    size_t n = (size_t)4 * 128 * 128 * 32;
    pool_gather_kernel<<<dim3((unsigned)((n + 255) / 256)), 256, 0, stream>>>(L1Ax, L1Ac, L2Ax, L2Ac, 256, 256);
  }
  nconv_wmma_kernel<1, false><<<g128, 256, 0, stream>>>(L2Ax, L2Ac, 0, 0, nullptr, nullptr, 0, 0,
                                                        wpD2, bd2, sD2, L2Bx, L2Bc, 128, 128);
  nconv_wmma_kernel<1, false><<<g128, 256, 0, stream>>>(L2Bx, L2Bc, 0, 0, nullptr, nullptr, 0, 0,
                                                        wpD3, bd3, sD3, L2Ax, L2Ac, 128, 128); // x2d,c2d
  // --- level 3 (64) ---
  {
    size_t n = (size_t)4 * 64 * 64 * 32;
    pool_gather_kernel<<<dim3((unsigned)((n + 255) / 256)), 256, 0, stream>>>(L2Ax, L2Ac, L3Ax, L3Ac, 128, 128);
  }
  nconv_wmma_kernel<1, false><<<g64, 256, 0, stream>>>(L3Ax, L3Ac, 0, 0, nullptr, nullptr, 0, 0,
                                                       wpD2, bd2, sD2, L3Bx, L3Bc, 64, 64);
  nconv_wmma_kernel<1, false><<<g64, 256, 0, stream>>>(L3Bx, L3Bc, 0, 0, nullptr, nullptr, 0, 0,
                                                       wpD3, bd3, sD3, L3Ax, L3Ac, 64, 64); // x3d,c3d
  // --- decoder ---
  // x34d = nconv(concat(x3d,x3d), concat(c3d,c3d), Wu1) @64
  nconv_wmma_kernel<2, false><<<g64, 256, 0, stream>>>(L3Ax, L3Ac, 0, 0, L3Ax, L3Ac, 0, 0,
                                                       wpU1, bu1, sU1, L3Bx, L3Bc, 64, 64);
  // x23d = nconv(concat(x2d, up2(x34d)), concat(c2d, up2(c34d)), Wu2) @128
  nconv_wmma_kernel<2, false><<<g128, 256, 0, stream>>>(L2Ax, L2Ac, 0, 0, L3Bx, L3Bc, 1, 1,
                                                        wpU2, bu2, sU2, L2Bx, L2Bc, 128, 128);
  // final: x-concat = (x1, up4(x23d)); c-concat = (up4(c23d), c1)  [swapped, faithful]
  float* coutp = (float*)d_out + (size_t)B * 3 * H * W;
  nconv_wmma_kernel<2, true><<<g512, 256, 0, stream>>>(F0x, L2Bc, 0, 2, L2Bx, F0c, 2, 0,
                                                       wpU3, bu3, sU3, Xpre, coutp, 512, 512);
  // fused 1x1 convs -> xout
  int HW = H * W;
  conv1x1_fused_kernel<<<dim3((B * HW + 255) / 256), 256, 0, stream>>>(
      Xpre, Wc1, bc1, Wc2, bc2, (float*)d_out, HW);
}
